// Network_1039382086437
// MI455X (gfx1250) — matrix operand order repaired and checked by reading, not compiled, not run on hardware
//
#include <hip/hip_runtime.h>

#define NTOT   (1024 * 128)   // total points
#define NMOD   4096           // voxel models (16^3)
#define WDIM   32

typedef __attribute__((ext_vector_type(16))) _Float16 v16h;
typedef __attribute__((ext_vector_type(8)))  _Float16 v8h;
typedef __attribute__((ext_vector_type(8)))  float    v8f;

// ---- WMMA fragment helpers (layouts per CDNA5 ISA §7.12.2) ----

// 16-bit A matrix 16x32: lanes 0-15 row M=lane hold K={0..7,16..23};
// lanes 16-31 row M=lane-16 hold K={8..15,24..31}.
__device__ __forceinline__ v16h ldsA(const _Float16* base, int stride, int lane) {
  int row = lane & 15;
  int kb  = (lane >> 4) << 3;                 // 0 or 8
  const _Float16* p = base + row * stride + kb;
  v8h lo = *(const v8h*)(p);
  v8h hi = *(const v8h*)(p + 16);
  v16h a;
#pragma unroll
  for (int i = 0; i < 8; ++i) { a[i] = lo[i]; a[i + 8] = hi[i]; }
  return a;
}

// 16-bit B matrix 32x16: lanes 0-15 col N=lane hold K={0..7,16..23};
// lanes 16-31 col N=lane-16 hold K={8..15,24..31}. Weights stored row-major w[out][in].
__device__ __forceinline__ v16h ldsB(const _Float16* w, int stride, int nh, int lane) {
  int o  = (lane & 15) + (nh << 4);
  int kb = (lane >> 4) << 3;                  // 0 or 8
  const _Float16* p = w + o * stride + kb;
  v8h lo = *(const v8h*)(p);
  v8h hi = *(const v8h*)(p + 16);
  v16h b;
#pragma unroll
  for (int i = 0; i < 8; ++i) { b[i] = lo[i]; b[i + 8] = hi[i]; }
  return b;
}

// f32 C/D 16x16: lane -> N=(lane&15)+16*nh, VGPR v -> M = v + 8*(lane>=16).
// Fuse bias (+ optional relu), convert to f16 and stage for the next layer's A.
__device__ __forceinline__ void stD(v8f d, _Float16* dst, int stride, int nh, int lane,
                                    const float* bias, bool do_relu) {
  int col = (lane & 15) + (nh << 4);
  int m0  = (lane >> 4) << 3;
  float b = bias[col];
#pragma unroll
  for (int v = 0; v < 8; ++v) {
    float val = d[v] + b;
    if (do_relu) val = fmaxf(val, 0.f);
    dst[(m0 + v) * stride + col] = (_Float16)val;
  }
}

__device__ __forceinline__ v8f wmma16(v16h a, v16h b, v8f c) {
  return __builtin_amdgcn_wmma_f32_16x16x32_f16(false, a, false, b, (short)0, c, false, false);
}

// ------------------------- bucketing kernels -----------------------------

__global__ void zero_kernel(int* __restrict__ counts) {
  int i = blockIdx.x * blockDim.x + threadIdx.x;
  if (i < NMOD) counts[i] = 0;
}

__global__ void idx_kernel(const float* __restrict__ pts, int* __restrict__ idxbuf,
                           int* __restrict__ counts) {
  int n = blockIdx.x * blockDim.x + threadIdx.x;
  if (n >= NTOT) return;
  float x = pts[n * 3 + 0], y = pts[n * 3 + 1], z = pts[n * 3 + 2];
  int ix = (int)fminf(fmaxf(x * 16.f, 0.f), 15.f);
  int iy = (int)fminf(fmaxf(y * 16.f, 0.f), 15.f);
  int iz = (int)fminf(fmaxf(z * 16.f, 0.f), 15.f);
  int id = ix * 256 + iy * 16 + iz;
  idxbuf[n] = id;
  atomicAdd(&counts[id], 1);
}

// One block, 1024 threads, 4 voxels each: exclusive prefix sum over 4096 counts.
__global__ __launch_bounds__(1024) void scan_kernel(const int* __restrict__ counts,
                                                    int* __restrict__ offsets,
                                                    int* __restrict__ cursor) {
  __shared__ int s[1024];
  int t = threadIdx.x;
  int c0 = counts[4 * t + 0], c1 = counts[4 * t + 1];
  int c2 = counts[4 * t + 2], c3 = counts[4 * t + 3];
  int sum = c0 + c1 + c2 + c3;
  s[t] = sum;
  __syncthreads();
  for (int off = 1; off < 1024; off <<= 1) {
    int v = (t >= off) ? s[t - off] : 0;
    __syncthreads();
    s[t] += v;
    __syncthreads();
  }
  int excl = s[t] - sum;
  int o0 = excl, o1 = o0 + c0, o2 = o1 + c1, o3 = o2 + c2;
  offsets[4 * t + 0] = o0; offsets[4 * t + 1] = o1;
  offsets[4 * t + 2] = o2; offsets[4 * t + 3] = o3;
  cursor[4 * t + 0] = o0;  cursor[4 * t + 1] = o1;
  cursor[4 * t + 2] = o2;  cursor[4 * t + 3] = o3;
}

__global__ void scatter_kernel(const int* __restrict__ idxbuf, int* __restrict__ cursor,
                               int* __restrict__ sorted) {
  int n = blockIdx.x * blockDim.x + threadIdx.x;
  if (n >= NTOT) return;
  int v = idxbuf[n];
  int pos = atomicAdd(&cursor[v], 1);
  sorted[pos] = n;
}

// --------------------------- per-voxel MLP -------------------------------

__global__ __launch_bounds__(128) void mlp_kernel(
    const float* __restrict__ pts, const float* __restrict__ vdirs,
    const int* __restrict__ sorted, const int* __restrict__ counts,
    const int* __restrict__ offsets,
    const float* __restrict__ w0, const float* __restrict__ b0,
    const float* __restrict__ w1, const float* __restrict__ b1,
    const float* __restrict__ fw, const float* __restrict__ fb,
    const float* __restrict__ sw, const float* __restrict__ sb,
    const float* __restrict__ vw, const float* __restrict__ vb,
    const float* __restrict__ rw, const float* __restrict__ rb,
    float* __restrict__ outRgb, float* __restrict__ outSig) {
  int vox = blockIdx.x;
  int cnt = counts[vox];
  if (cnt == 0) return;                     // block-uniform early exit
  int base = offsets[vox];
  int tid = threadIdx.x, lane = tid & 31, wv = tid >> 5;

  // f16 weight tiles (zero-padded to K=64 where needed), f32 biases.
  __shared__ __align__(32) _Float16 w0h[WDIM * 64];   // 63 -> 64
  __shared__ __align__(32) _Float16 w1h[WDIM * WDIM];
  __shared__ __align__(32) _Float16 fwh[WDIM * WDIM];
  __shared__ __align__(32) _Float16 vwh[WDIM * 64];   // 59 -> 64
  __shared__ float b0s[WDIM], b1s[WDIM], fbs[WDIM], vbs[WDIM], sws[WDIM], rws[96];
  __shared__ float sbs, rbs[3];
  // per-wave activation staging
  __shared__ __align__(32) _Float16 eph[4][16 * 64];  // xyz embedding (63 + pad)
  __shared__ __align__(32) _Float16 hbf[4][16 * 32];  // hidden
  __shared__ __align__(32) _Float16 h2b[4][16 * 64];  // [feat(32) | dir-embed(27) | pad]

  for (int i = tid; i < WDIM * 64; i += 128) {
    int o = i >> 6, k = i & 63;
    w0h[i] = (_Float16)(k < 63 ? w0[(size_t)vox * (WDIM * 63) + o * 63 + k] : 0.f);
    vwh[i] = (_Float16)(k < 59 ? vw[(size_t)vox * (WDIM * 59) + o * 59 + k] : 0.f);
  }
  for (int i = tid; i < WDIM * WDIM; i += 128) {
    w1h[i] = (_Float16)w1[(size_t)vox * (WDIM * WDIM) + i];
    fwh[i] = (_Float16)fw[(size_t)vox * (WDIM * WDIM) + i];
  }
  if (tid < WDIM) {
    b0s[tid] = b0[vox * WDIM + tid];
    b1s[tid] = b1[vox * WDIM + tid];
    fbs[tid] = fb[vox * WDIM + tid];
    vbs[tid] = vb[vox * WDIM + tid];
    sws[tid] = sw[vox * WDIM + tid];
  }
  if (tid < 96) rws[tid] = rw[vox * 96 + tid];
  if (tid == 0) sbs = sb[vox];
  if (tid < 3)  rbs[tid] = rb[vox * 3 + tid];
  __syncthreads();

  _Float16* EP = eph[wv];
  _Float16* HB = hbf[wv];
  _Float16* H2 = h2b[wv];

  int ntiles = (cnt + 15) >> 4;
  int passes = (ntiles + 3) >> 2;           // uniform trip count for all 4 waves

  for (int tp = 0; tp < passes; ++tp) {
    int tile = tp * 4 + wv;
    bool act = tile < ntiles;               // wave-uniform -> EXEC all-ones for WMMA
    int m = lane & 15;
    int pofs = tile * 16 + m;
    bool valid = act && (pofs < cnt);
    int gid = sorted[valid ? (base + pofs) : base];

    if (act) {  // stage embeddings: lanes 0-15 xyz (L=10), lanes 16-31 dir (L=4)
      if (lane < 16) {
        float c3[3] = {pts[gid * 3 + 0], pts[gid * 3 + 1], pts[gid * 3 + 2]};
        _Float16* row = EP + m * 64;
        row[0] = (_Float16)c3[0]; row[1] = (_Float16)c3[1]; row[2] = (_Float16)c3[2];
        float fr = 1.f;
#pragma unroll
        for (int k = 0; k < 10; ++k) {
#pragma unroll
          for (int c = 0; c < 3; ++c) {
            float sn, cs; __sincosf(c3[c] * fr, &sn, &cs);
            row[3 + k * 6 + c]     = (_Float16)sn;
            row[3 + k * 6 + 3 + c] = (_Float16)cs;
          }
          fr *= 2.f;
        }
        row[63] = (_Float16)0.f;
      } else {
        int ray = gid >> 7;                 // N_SAMPLES = 128
        float c3[3] = {vdirs[ray * 3 + 0], vdirs[ray * 3 + 1], vdirs[ray * 3 + 2]};
        _Float16* row = H2 + m * 64 + 32;
        row[0] = (_Float16)c3[0]; row[1] = (_Float16)c3[1]; row[2] = (_Float16)c3[2];
        float fr = 1.f;
#pragma unroll
        for (int k = 0; k < 4; ++k) {
#pragma unroll
          for (int c = 0; c < 3; ++c) {
            float sn, cs; __sincosf(c3[c] * fr, &sn, &cs);
            row[3 + k * 6 + c]     = (_Float16)sn;
            row[3 + k * 6 + 3 + c] = (_Float16)cs;
          }
          fr *= 2.f;
        }
#pragma unroll
        for (int j = 27; j < 32; ++j) row[j] = (_Float16)0.f;
      }
    }
    __syncthreads();

    if (act) {  // layer0: ep(16x64) @ w0^T -> relu -> HB(16x32)
      v8f d0 = {}, d1 = {};
#pragma unroll
      for (int kc = 0; kc < 2; ++kc) {
        v16h a = ldsA(EP + kc * 32, 64, lane);
        d0 = wmma16(a, ldsB(w0h + kc * 32, 64, 0, lane), d0);
        d1 = wmma16(a, ldsB(w0h + kc * 32, 64, 1, lane), d1);
      }
      stD(d0, HB, 32, 0, lane, b0s, true);
      stD(d1, HB, 32, 1, lane, b0s, true);
    }
    __syncthreads();

    if (act) {  // layer1: HB @ w1^T -> relu -> HB (in-place; DS ops in-order per wave)
      v16h a = ldsA(HB, 32, lane);
      v8f d0 = {}, d1 = {};
      d0 = wmma16(a, ldsB(w1h, 32, 0, lane), d0);
      d1 = wmma16(a, ldsB(w1h, 32, 1, lane), d1);
      stD(d0, HB, 32, 0, lane, b1s, true);
      stD(d1, HB, 32, 1, lane, b1s, true);
    }
    __syncthreads();

    if (act) {  // feat (no relu) -> H2[:, :32]; sigma = h . sigma_w via VALU
      v16h a = ldsA(HB, 32, lane);
      v8f d0 = {}, d1 = {};
      d0 = wmma16(a, ldsB(fwh, 32, 0, lane), d0);
      d1 = wmma16(a, ldsB(fwh, 32, 1, lane), d1);
      stD(d0, H2, 64, 0, lane, fbs, false);
      stD(d1, H2, 64, 1, lane, fbs, false);
      if (lane < 16 && valid) {
        float s = sbs;
#pragma unroll
        for (int k = 0; k < 32; ++k) s += (float)HB[m * 32 + k] * sws[k];
        outSig[gid] = s;
      }
    }
    __syncthreads();

    if (act) {  // view: H2(16x64) @ vw^T -> relu -> HB
      v8f d0 = {}, d1 = {};
#pragma unroll
      for (int kc = 0; kc < 2; ++kc) {
        v16h a = ldsA(H2 + kc * 32, 64, lane);
        d0 = wmma16(a, ldsB(vwh + kc * 32, 64, 0, lane), d0);
        d1 = wmma16(a, ldsB(vwh + kc * 32, 64, 1, lane), d1);
      }
      stD(d0, HB, 32, 0, lane, vbs, true);
      stD(d1, HB, 32, 1, lane, vbs, true);
    }
    __syncthreads();

    if (act && lane < 16 && valid) {  // rgb: 32 -> 3 via VALU
      float r0 = rbs[0], r1 = rbs[1], r2 = rbs[2];
#pragma unroll
      for (int k = 0; k < 32; ++k) {
        float hv = (float)HB[m * 32 + k];
        r0 += hv * rws[k]; r1 += hv * rws[32 + k]; r2 += hv * rws[64 + k];
      }
      outRgb[gid * 3 + 0] = r0;
      outRgb[gid * 3 + 1] = r1;
      outRgb[gid * 3 + 2] = r2;
    }
    __syncthreads();
  }
}

// ------------------------------ launcher ---------------------------------

extern "C" void kernel_launch(void* const* d_in, const int* in_sizes, int n_in,
                              void* d_out, int out_size, void* d_ws, size_t ws_size,
                              hipStream_t stream) {
  (void)in_sizes; (void)n_in; (void)out_size; (void)ws_size;
  const float* pts  = (const float*)d_in[0];
  const float* vdir = (const float*)d_in[1];
  const float* w0   = (const float*)d_in[2];
  const float* b0   = (const float*)d_in[3];
  const float* w1   = (const float*)d_in[4];
  const float* b1   = (const float*)d_in[5];
  const float* fw   = (const float*)d_in[6];
  const float* fb   = (const float*)d_in[7];
  const float* sw   = (const float*)d_in[8];
  const float* sb   = (const float*)d_in[9];
  const float* vw   = (const float*)d_in[10];
  const float* vb   = (const float*)d_in[11];
  const float* rw   = (const float*)d_in[12];
  const float* rb   = (const float*)d_in[13];

  // tuple output: rgb (N*3 floats) then sigma (N floats)
  float* outRgb = (float*)d_out;
  float* outSig = outRgb + (size_t)NTOT * 3;

  // workspace: ~1.05 MB of int scratch
  int* idxbuf  = (int*)d_ws;          // NTOT
  int* counts  = idxbuf + NTOT;       // NMOD
  int* offsets = counts + NMOD;       // NMOD
  int* cursor  = offsets + NMOD;      // NMOD
  int* sorted  = cursor + NMOD;       // NTOT

  zero_kernel<<<(NMOD + 255) / 256, 256, 0, stream>>>(counts);
  idx_kernel<<<(NTOT + 255) / 256, 256, 0, stream>>>(pts, idxbuf, counts);
  scan_kernel<<<1, 1024, 0, stream>>>(counts, offsets, cursor);
  scatter_kernel<<<(NTOT + 255) / 256, 256, 0, stream>>>(idxbuf, cursor, sorted);
  mlp_kernel<<<NMOD, 128, 0, stream>>>(pts, vdir, sorted, counts, offsets,
                                       w0, b0, w1, b1, fw, fb, sw, sb,
                                       vw, vb, rw, rb, outRgb, outSig);
}
